// ValkyrieModel_63728724738309
// MI455X (gfx1250) — matrix operand order repaired and checked by reading, not compiled, not run on hardware
//
#include <hip/hip_runtime.h>
#include <cstddef>
#include <cstdint>

// ---------------- model constants ----------------
constexpr int VOC  = 32000;
constexpr int Dm   = 1024;
constexpr int NH   = 16;
constexpr int HDim = 64;
constexpr int NL   = 4;
constexpr int SEQ  = 2048;
constexpr int BAT  = 2;
constexpr int T    = BAT * SEQ;     // 4096 tokens
constexpr int HID  = 2736;
constexpr int HIDp = 2752;          // padded to multiple of 32 for WMMA K-loop
constexpr float EPS = 1e-6f;

// ---------------- WMMA types ----------------
typedef __attribute__((ext_vector_type(16))) __bf16 v16bf;
typedef __attribute__((ext_vector_type(8)))  __bf16 v8bf;
typedef __attribute__((ext_vector_type(8)))  float  v8f;

union FragU { v16bf v; v8bf h[2]; };

static __device__ __forceinline__ v8f wmma_bf16(v16bf a, v16bf b, v8f c) {
  return __builtin_amdgcn_wmma_f32_16x16x32_bf16(false, a, false, b, (short)0, c, false, false);
}

static __device__ __forceinline__ v8bf zero_v8bf() {
  union { unsigned u[4]; v8bf v; } z;
  z.u[0] = z.u[1] = z.u[2] = z.u[3] = 0u;
  return z.v;
}

// ---------------- TDM (Tensor Data Mover) support ----------------
#if defined(__has_builtin)
#if __has_builtin(__builtin_amdgcn_tensor_load_to_lds) && __has_builtin(__builtin_amdgcn_s_wait_tensorcnt)
#define USE_TDM 1
#endif
#endif

#ifdef USE_TDM
typedef __attribute__((ext_vector_type(4))) unsigned u32x4;
typedef __attribute__((ext_vector_type(8))) int      i32x8;
typedef __attribute__((ext_vector_type(4))) int      i32x4;

// LDS byte offset of a __shared__ object (addrspacecast generic->as3, ptrtoint)
static __device__ __forceinline__ unsigned lds_offset(void* p) {
  return (unsigned)(unsigned long long)(__attribute__((address_space(3))) char*)p;
}

// DMA a 2D tile (32 bf16 cols x 128 rows, row stride = K elements) into LDS with
// 16B padding after each 64B row (=> 80B LDS row stride, matches fragment reads).
// Rows beyond `rows` are OOB -> read as zero (handles N-tail of Bt).
static __device__ __forceinline__ void tdm_load_tile(
    unsigned lds_off, const __bf16* gtile, int rows, int K) {
  const unsigned long long ga = (unsigned long long)(const void*)gtile;
  u32x4 g0;
  g0[0] = 1u;                                              // count=1, user mode
  g0[1] = lds_off;                                         // lds_addr (bytes)
  g0[2] = (unsigned)(ga & 0xFFFFFFFFu);                    // global_addr[31:0]
  g0[3] = (unsigned)((ga >> 32) & 0x01FFFFFFu) | (2u << 30); // addr[56:32] | type=2
  i32x8 g1;
  g1[0] = (int)(0x00010000u        // data_size = 1 (2 bytes)
              | (1u << 20)         // pad_enable
              | (3u << 22)         // pad_interval: 16 DWORDs (64B row)
              | (3u << 25));       // pad_amount:   4 DWORDs (16B pad)
  g1[1] = (int)(((unsigned)K & 0xFFFFu) << 16);            // tensor_dim0[15:0]
  g1[2] = (int)(((unsigned)K >> 16) |
                (((unsigned)rows & 0xFFFFu) << 16));       // dim0 hi | tensor_dim1 lo
  g1[3] = (int)(((unsigned)rows >> 16) | (32u << 16));     // dim1 hi | tile_dim0=32
  g1[4] = 128;                                             // tile_dim1 = 128 rows
  g1[5] = K;                                               // tensor_dim0_stride lo
  g1[6] = 0;
  g1[7] = 0;
  i32x4 z4 = {};
  i32x8 z8 = {};
  // 6-arg form (clang-23 / therock-10.0 headers): groups 2,3 + extra group, cpol
  __builtin_amdgcn_tensor_load_to_lds(g0, g1, z4, z4, z8, 0);
}
#endif  // USE_TDM

// ---------------- converters ----------------
// f32 -> bf16 flat copy (embedding: already [N=V, K=D] row-major for logits GEMM)
__global__ __launch_bounds__(256) void convert_bf16_kernel(
    const float* __restrict__ src, __bf16* __restrict__ dst, size_t n) {
  size_t i = ((size_t)blockIdx.x * 256 + threadIdx.x) * 4;
  if (i + 3 < n) {
    float4 v = *(const float4*)(src + i);
    dst[i + 0] = (__bf16)v.x;
    dst[i + 1] = (__bf16)v.y;
    dst[i + 2] = (__bf16)v.z;
    dst[i + 3] = (__bf16)v.w;
  }
}

// src[K][N] f32 -> dst[N][Kpad] bf16 (transpose; zero-fill K..Kpad)
__global__ __launch_bounds__(256) void transpose_convert_kernel(
    const float* __restrict__ src, __bf16* __restrict__ dst,
    int K, int N, int Kpad) {
  __shared__ float tile[32][33];
  const int n0 = blockIdx.x * 32, k0 = blockIdx.y * 32;
#pragma unroll
  for (int i = 0; i < 4; ++i) {
    int kk = k0 + threadIdx.y + i * 8;
    int nn = n0 + threadIdx.x;
    tile[threadIdx.y + i * 8][threadIdx.x] =
        (kk < K && nn < N) ? src[(size_t)kk * N + nn] : 0.f;
  }
  __syncthreads();
#pragma unroll
  for (int i = 0; i < 4; ++i) {
    int nn = n0 + threadIdx.y + i * 8;
    int kk = k0 + threadIdx.x;
    if (nn < N && kk < Kpad)
      dst[(size_t)nn * Kpad + kk] = (__bf16)tile[threadIdx.x][threadIdx.y + i * 8];
  }
}

// ---------------- embedding gather ----------------
__global__ __launch_bounds__(256) void embed_kernel(
    const int* __restrict__ ids, const float* __restrict__ emb,
    float* __restrict__ x) {
  const int t = blockIdx.x;
  const int id = ids[t];
  const float4* src = (const float4*)(emb + (size_t)id * Dm);
  float4* dst = (float4*)(x + (size_t)t * Dm);
  dst[threadIdx.x] = src[threadIdx.x];
}

// ---------------- RMSNorm: f32 x -> bf16 xn ----------------
__global__ __launch_bounds__(256) void rmsnorm_kernel(
    const float* __restrict__ x, const float* __restrict__ w,
    __bf16* __restrict__ out) {
  const int row = blockIdx.x;
  const float* xr = x + (size_t)row * Dm;
  float4 v = ((const float4*)xr)[threadIdx.x];
  float s = v.x * v.x + v.y * v.y + v.z * v.z + v.w * v.w;
#pragma unroll
  for (int o = 16; o >= 1; o >>= 1) s += __shfl_xor(s, o, 32);
  __shared__ float red[8];
  if ((threadIdx.x & 31) == 0) red[threadIdx.x >> 5] = s;
  __syncthreads();
  float tot = 0.f;
#pragma unroll
  for (int i = 0; i < 8; ++i) tot += red[i];
  const float inv = rsqrtf(tot * (1.f / (float)Dm) + EPS);
  const float4 wv = ((const float4*)w)[threadIdx.x];
  const int j = threadIdx.x * 4;
  __bf16* o2 = out + (size_t)row * Dm + j;
  o2[0] = (__bf16)(v.x * inv * wv.x);
  o2[1] = (__bf16)(v.y * inv * wv.y);
  o2[2] = (__bf16)(v.z * inv * wv.z);
  o2[3] = (__bf16)(v.w * inv * wv.w);
}

// ---------------- RoPE (in-place on bf16 [T, D]) ----------------
__global__ __launch_bounds__(256) void rope_kernel(__bf16* __restrict__ qk) {
  const int t = blockIdx.x;
  const int idx = blockIdx.y * 256 + threadIdx.x;   // 0..511 -> (h, j)
  const int h = idx >> 5, j = idx & 31;
  const int p = t & (SEQ - 1);
  const size_t base = (size_t)t * Dm + h * HDim + j;
  const float x1 = (float)qk[base];
  const float x2 = (float)qk[base + 32];
  // inv_freq = 10000^(-2j/64) = exp(-j * ln(10000)/32)
  const float invf = __expf(-(float)j * 0.28782313662425572f);
  const float ang = (float)p * invf;
  const float c = __cosf(ang), s = __sinf(ang);
  qk[base]      = (__bf16)(x1 * c - x2 * s);
  qk[base + 32] = (__bf16)(x1 * s + x2 * c);
}

// ---------------- SiLU(g)*u -> padded bf16 h ----------------
__global__ __launch_bounds__(256) void silu_mul_kernel(
    const float* __restrict__ g, const float* __restrict__ u,
    __bf16* __restrict__ h) {
  const size_t i = (size_t)blockIdx.x * 256 + threadIdx.x;
  if (i >= (size_t)T * HIDp) return;
  const int t = (int)(i / HIDp);
  const int j = (int)(i - (size_t)t * HIDp);
  float val = 0.f;
  if (j < HID) {
    const float gv = g[(size_t)t * HID + j];
    const float uv = u[(size_t)t * HID + j];
    val = gv / (1.f + __expf(-gv)) * uv;
  }
  h[i] = (__bf16)val;
}

// ---------------- WMMA GEMM: C[M,N] = A[M,K] * Bt[N,K]^T ----------------
enum { EPI_BF16 = 0, EPI_F32 = 1, EPI_RESID = 2 };

template <int EPI>
__global__ __launch_bounds__(256) void gemm_kernel(
    const __bf16* __restrict__ A, const __bf16* __restrict__ Bt,
    int M, int N, int K,
    float* __restrict__ outF, __bf16* __restrict__ outB,
    const float* __restrict__ resid) {
  constexpr int LDT = 40;  // halves per LDS row: 80B, 16B-aligned, conflict-free
  __shared__ __attribute__((aligned(16))) __bf16 lA[2][128 * LDT];
  __shared__ __attribute__((aligned(16))) __bf16 lB[2][128 * LDT];
  const int tid = threadIdx.x, wid = tid >> 5, lane = tid & 31;
  const int lrow = lane & 15, lhi = lane >> 4;
  const int wm = wid >> 2, wn = wid & 3;   // 2x4 wave grid
  const int bm = blockIdx.y * 128, bn = blockIdx.x * 128;
  v8f acc[4][2] = {};
  const int nsteps = K / 32;

#ifdef USE_TDM
  // -------- TDM double-buffered pipeline: DMA tiles, WMMA never stalls -------
  if (tid < 32) {  // wave 0 drives the Tensor Data Mover
    tdm_load_tile(lds_offset(&lA[0][0]), A  + (size_t)bm * K, M - bm, K);
    tdm_load_tile(lds_offset(&lB[0][0]), Bt + (size_t)bn * K, N - bn, K);
  }
#pragma unroll 1
  for (int s = 0; s < nsteps; ++s) {
    const int buf = s & 1;
    if (s + 1 < nsteps) {
      if (tid < 32) {
        tdm_load_tile(lds_offset(&lA[buf ^ 1][0]),
                      A  + (size_t)bm * K + (s + 1) * 32, M - bm, K);
        tdm_load_tile(lds_offset(&lB[buf ^ 1][0]),
                      Bt + (size_t)bn * K + (s + 1) * 32, N - bn, K);
        __builtin_amdgcn_s_wait_tensorcnt(2);  // current buffer's 2 DMAs done
      }
    } else {
      if (tid < 32) __builtin_amdgcn_s_wait_tensorcnt(0);
    }
    __syncthreads();  // current tiles visible to all waves

    FragU af[4], bfr[2];
#pragma unroll
    for (int mi = 0; mi < 4; ++mi) {
      const __bf16* p = &lA[buf][(wm * 64 + mi * 16 + lrow) * LDT];
      const int base = lhi * 8;
      af[mi].h[0] = *(const v8bf*)(p + base);
      af[mi].h[1] = *(const v8bf*)(p + base + 16);
    }
#pragma unroll
    for (int ni = 0; ni < 2; ++ni) {
      const __bf16* p = &lB[buf][(wn * 32 + ni * 16 + lrow) * LDT + lhi * 8];
      bfr[ni].h[0] = *(const v8bf*)p;
      bfr[ni].h[1] = *(const v8bf*)(p + 16);
    }
#pragma unroll
    for (int mi = 0; mi < 4; ++mi)
#pragma unroll
      for (int ni = 0; ni < 2; ++ni)
        acc[mi][ni] = wmma_bf16(af[mi].v, bfr[ni].v, acc[mi][ni]);

    __syncthreads();  // all waves done reading buf before it is re-filled
  }
#else
  // -------- fallback: cooperative VMEM staging (host pass / old toolchains) --
  const int arow = tid >> 1, aoff = (tid & 1) * 16;
#pragma unroll 1
  for (int s = 0; s < nsteps; ++s) {
    const int k0 = s * 32;
    __syncthreads();
    {
      const __bf16* ap = A + (size_t)(bm + arow) * K + k0 + aoff;
      *(v8bf*)&lA[0][arow * LDT + aoff]     = *(const v8bf*)ap;
      *(v8bf*)&lA[0][arow * LDT + aoff + 8] = *(const v8bf*)(ap + 8);
    }
    {
      const int nrow = bn + arow;
      if (nrow < N) {
        const __bf16* bp = Bt + (size_t)nrow * K + k0 + aoff;
        *(v8bf*)&lB[0][arow * LDT + aoff]     = *(const v8bf*)bp;
        *(v8bf*)&lB[0][arow * LDT + aoff + 8] = *(const v8bf*)(bp + 8);
      } else {
        v8bf z = zero_v8bf();
        *(v8bf*)&lB[0][arow * LDT + aoff]     = z;
        *(v8bf*)&lB[0][arow * LDT + aoff + 8] = z;
      }
    }
    __syncthreads();
    FragU af[4], bfr[2];
#pragma unroll
    for (int mi = 0; mi < 4; ++mi) {
      const __bf16* p = &lA[0][(wm * 64 + mi * 16 + lrow) * LDT];
      const int base = lhi * 8;
      af[mi].h[0] = *(const v8bf*)(p + base);
      af[mi].h[1] = *(const v8bf*)(p + base + 16);
    }
#pragma unroll
    for (int ni = 0; ni < 2; ++ni) {
      const __bf16* p = &lB[0][(wn * 32 + ni * 16 + lrow) * LDT + lhi * 8];
      bfr[ni].h[0] = *(const v8bf*)p;
      bfr[ni].h[1] = *(const v8bf*)(p + 16);
    }
#pragma unroll
    for (int mi = 0; mi < 4; ++mi)
#pragma unroll
      for (int ni = 0; ni < 2; ++ni)
        acc[mi][ni] = wmma_bf16(af[mi].v, bfr[ni].v, acc[mi][ni]);
  }
#endif

  // epilogue: C layout -> VGPR i holds row (i + 8*lhi), column = lane&15
#pragma unroll
  for (int mi = 0; mi < 4; ++mi)
#pragma unroll
    for (int ni = 0; ni < 2; ++ni)
#pragma unroll
      for (int i = 0; i < 8; ++i) {
        const int row = bm + wm * 64 + mi * 16 + i + 8 * lhi;
        const int col = bn + wn * 32 + ni * 16 + lrow;
        if (col < N) {
          const size_t idx = (size_t)row * N + col;
          const float vv = acc[mi][ni][i];
          if (EPI == EPI_BF16)      outB[idx] = (__bf16)vv;
          else if (EPI == EPI_F32)  outF[idx] = vv;
          else                      outF[idx] = resid[idx] + vv;
        }
      }
}

// ---------------- blocked attention (flash-style over 3 key blocks) ----------
__global__ __launch_bounds__(128) void attention_kernel(
    const __bf16* __restrict__ q, const __bf16* __restrict__ k,
    const __bf16* __restrict__ v, __bf16* __restrict__ out) {
  __shared__ __attribute__((aligned(16))) __bf16 vt[64 * 72];      // V^T [hd][key]
  __shared__ __attribute__((aligned(16))) __bf16 pl[4][16 * 72];   // per-wave P tile
  const int blk = blockIdx.x, h = blockIdx.y, b = blockIdx.z;
  const int tid = threadIdx.x, wid = tid >> 5, lane = tid & 31;
  const int lrow = lane & 15, lhi = lane >> 4;
  const int t0 = b * SEQ + blk * 64;

  // Q A-fragments (hd 0..31 and 32..63)
  FragU a0, a1;
  {
    const __bf16* qp = q + (size_t)(t0 + wid * 16 + lrow) * Dm + h * HDim;
    const int base = lhi * 8;
    a0.h[0] = *(const v8bf*)(qp + base);
    a0.h[1] = *(const v8bf*)(qp + base + 16);
    a1.h[0] = *(const v8bf*)(qp + 32 + base);
    a1.h[1] = *(const v8bf*)(qp + 32 + base + 16);
  }

  float m_[8], l_[8];
  v8f o_[4] = {};
#pragma unroll
  for (int i = 0; i < 8; ++i) { m_[i] = -3.0e38f; l_[i] = 0.f; }

#pragma unroll 1
  for (int kb = 0; kb < 3; ++kb) {
    if (kb == 1 && blk < 2) continue;   // uniform per block
    if (kb == 2 && blk < 1) continue;
    const int kbi = (kb == 0) ? 0 : ((kb == 1) ? (blk - 1) : blk);
    const int kt0 = b * SEQ + kbi * 64;

    __syncthreads();  // protect vt from previous iteration's readers
    {  // stage V^T: each of 128 threads copies 32 contiguous halves of one V row
      const int key = tid >> 1, hoff = (tid & 1) * 32;
      const __bf16* vp = v + (size_t)(kt0 + key) * Dm + h * HDim + hoff;
      v8bf tmp[4];
#pragma unroll
      for (int c = 0; c < 4; ++c) tmp[c] = *(const v8bf*)(vp + c * 8);
#pragma unroll
      for (int c = 0; c < 4; ++c)
#pragma unroll
        for (int j = 0; j < 8; ++j) vt[(hoff + c * 8 + j) * 72 + key] = tmp[c][j];
    }
    __syncthreads();

    // scores S = Q * K^T (16x64 per wave, 4 N-tiles, K-dim = HD chained 2x32)
    v8f s_[4];
#pragma unroll
    for (int nt = 0; nt < 4; ++nt) {
      FragU b0, b1;
      const __bf16* kp = k + (size_t)(kt0 + nt * 16 + lrow) * Dm + h * HDim + lhi * 8;
      b0.h[0] = *(const v8bf*)(kp);      b0.h[1] = *(const v8bf*)(kp + 16);
      b1.h[0] = *(const v8bf*)(kp + 32); b1.h[1] = *(const v8bf*)(kp + 48);
      v8f zacc = {};
      zacc = wmma_bf16(a0.v, b0.v, zacc);
      s_[nt] = wmma_bf16(a1.v, b1.v, zacc);
    }

    // online softmax update
#pragma unroll
    for (int i = 0; i < 8; ++i) {
      const int qpos = blk * 64 + wid * 16 + i + 8 * lhi;
      float mx = -3.0e38f;
#pragma unroll
      for (int nt = 0; nt < 4; ++nt) {
        const int kpos = kbi * 64 + nt * 16 + lrow;
        float val = (kpos <= qpos) ? s_[nt][i] * 0.125f : -3.0e38f;
        s_[nt][i] = val;
        mx = fmaxf(mx, val);
      }
#pragma unroll
      for (int off = 1; off < 16; off <<= 1) mx = fmaxf(mx, __shfl_xor(mx, off, 16));
      const float mnew = fmaxf(m_[i], mx);
      const float f = __expf(m_[i] - mnew);
      float rsum = 0.f;
#pragma unroll
      for (int nt = 0; nt < 4; ++nt) {
        const float p = __expf(s_[nt][i] - mnew);
        s_[nt][i] = p;
        rsum += p;
      }
#pragma unroll
      for (int off = 1; off < 16; off <<= 1) rsum += __shfl_xor(rsum, off, 16);
      l_[i] = l_[i] * f + rsum;
      m_[i] = mnew;
#pragma unroll
      for (int nt = 0; nt < 4; ++nt) o_[nt][i] *= f;
    }

    // stage P (C-layout) into per-wave LDS, reload in A-layout
    __bf16* pw = &pl[wid][0];
#pragma unroll
    for (int i = 0; i < 8; ++i)
#pragma unroll
      for (int nt = 0; nt < 4; ++nt)
        pw[(i + 8 * lhi) * 72 + nt * 16 + lrow] = (__bf16)s_[nt][i];
    asm volatile("s_wait_dscnt 0" ::: "memory");

    FragU pa0, pa1;
    {
      const __bf16* pp = pw + lrow * 72;
      const int base = lhi * 8;
      pa0.h[0] = *(const v8bf*)(pp + base);
      pa0.h[1] = *(const v8bf*)(pp + base + 16);
      pa1.h[0] = *(const v8bf*)(pp + 32 + base);
      pa1.h[1] = *(const v8bf*)(pp + 32 + base + 16);
    }

    // O += P * V  (B fragments from V^T in LDS: contiguous over keys)
#pragma unroll
    for (int nt = 0; nt < 4; ++nt) {
      FragU vb0, vb1;
      const __bf16* vp2 = &vt[(nt * 16 + lrow) * 72 + lhi * 8];
      vb0.h[0] = *(const v8bf*)(vp2);      vb0.h[1] = *(const v8bf*)(vp2 + 16);
      vb1.h[0] = *(const v8bf*)(vp2 + 32); vb1.h[1] = *(const v8bf*)(vp2 + 48);
      v8f tacc = wmma_bf16(pa0.v, vb0.v, o_[nt]);
      o_[nt] = wmma_bf16(pa1.v, vb1.v, tacc);
    }
  }

  // normalize and store
#pragma unroll
  for (int nt = 0; nt < 4; ++nt)
#pragma unroll
    for (int i = 0; i < 8; ++i) {
      const int row = wid * 16 + i + 8 * lhi;
      const int col = nt * 16 + lrow;
      out[(size_t)(t0 + row) * Dm + h * HDim + col] = (__bf16)(o_[nt][i] / l_[i]);
    }
}

// ---------------- host orchestration ----------------
extern "C" void kernel_launch(void* const* d_in, const int* in_sizes, int n_in,
                              void* d_out, int out_size, void* d_ws, size_t ws_size,
                              hipStream_t stream) {
  (void)in_sizes; (void)n_in; (void)out_size; (void)ws_size;
  const int*   ids = (const int*)d_in[0];
  const float* emb = (const float*)d_in[1];
  const float* Wq  = (const float*)d_in[2];
  const float* Wk  = (const float*)d_in[3];
  const float* Wv  = (const float*)d_in[4];
  const float* Wo  = (const float*)d_in[5];
  const float* Wg  = (const float*)d_in[6];
  const float* Wu  = (const float*)d_in[7];
  const float* Wd  = (const float*)d_in[8];
  const float* n1w = (const float*)d_in[9];
  const float* n2w = (const float*)d_in[10];
  const float* fnw = (const float*)d_in[11];
  float* logits = (float*)d_out;

  char* base = (char*)d_ws;
  size_t off = 0;
  auto take = [&](size_t bytes) -> char* {
    char* p = base + off;
    off += (bytes + 255) & ~(size_t)255;
    return p;
  };

  float*  x    = (float*) take((size_t)T * Dm * 4);
  __bf16* xn   = (__bf16*)take((size_t)T * Dm * 2);
  __bf16* qb   = (__bf16*)take((size_t)T * Dm * 2);
  __bf16* kb   = (__bf16*)take((size_t)T * Dm * 2);
  __bf16* vb   = (__bf16*)take((size_t)T * Dm * 2);
  __bf16* ab   = (__bf16*)take((size_t)T * Dm * 2);
  float*  gbuf = (float*) take((size_t)T * HID * 4);
  float*  ubuf = (float*) take((size_t)T * HID * 4);
  __bf16* hp   = (__bf16*)take((size_t)T * HIDp * 2);
  __bf16* embB = (__bf16*)take((size_t)VOC * Dm * 2);
  __bf16* WqT  = (__bf16*)take((size_t)NL * Dm * Dm * 2);
  __bf16* WkT  = (__bf16*)take((size_t)NL * Dm * Dm * 2);
  __bf16* WvT  = (__bf16*)take((size_t)NL * Dm * Dm * 2);
  __bf16* WoT  = (__bf16*)take((size_t)NL * Dm * Dm * 2);
  __bf16* WgT  = (__bf16*)take((size_t)NL * HID * Dm * 2);
  __bf16* WuT  = (__bf16*)take((size_t)NL * HID * Dm * 2);
  __bf16* WdT  = (__bf16*)take((size_t)NL * Dm * HIDp * 2);

  const dim3 tcb(32, 8);

  // --- precision conversion (bf16) + weight pre-transposition ---
  convert_bf16_kernel<<<dim3(((size_t)VOC * Dm) / 1024), 256, 0, stream>>>(
      emb, embB, (size_t)VOC * Dm);
  for (int l = 0; l < NL; ++l) {
    transpose_convert_kernel<<<dim3(Dm / 32, Dm / 32), tcb, 0, stream>>>(
        Wq + (size_t)l * Dm * Dm, WqT + (size_t)l * Dm * Dm, Dm, Dm, Dm);
    transpose_convert_kernel<<<dim3(Dm / 32, Dm / 32), tcb, 0, stream>>>(
        Wk + (size_t)l * Dm * Dm, WkT + (size_t)l * Dm * Dm, Dm, Dm, Dm);
    transpose_convert_kernel<<<dim3(Dm / 32, Dm / 32), tcb, 0, stream>>>(
        Wv + (size_t)l * Dm * Dm, WvT + (size_t)l * Dm * Dm, Dm, Dm, Dm);
    transpose_convert_kernel<<<dim3(Dm / 32, Dm / 32), tcb, 0, stream>>>(
        Wo + (size_t)l * Dm * Dm, WoT + (size_t)l * Dm * Dm, Dm, Dm, Dm);
    transpose_convert_kernel<<<dim3((HID + 31) / 32, Dm / 32), tcb, 0, stream>>>(
        Wg + (size_t)l * Dm * HID, WgT + (size_t)l * HID * Dm, Dm, HID, Dm);
    transpose_convert_kernel<<<dim3((HID + 31) / 32, Dm / 32), tcb, 0, stream>>>(
        Wu + (size_t)l * Dm * HID, WuT + (size_t)l * HID * Dm, Dm, HID, Dm);
    transpose_convert_kernel<<<dim3(Dm / 32, HIDp / 32), tcb, 0, stream>>>(
        Wd + (size_t)l * HID * Dm, WdT + (size_t)l * Dm * HIDp, HID, Dm, HIDp);
  }

  // --- embedding gather ---
  embed_kernel<<<dim3(T), 256, 0, stream>>>(ids, emb, x);

  // --- transformer layers ---
  for (int l = 0; l < NL; ++l) {
    rmsnorm_kernel<<<dim3(T), 256, 0, stream>>>(x, n1w + (size_t)l * Dm, xn);

    gemm_kernel<EPI_BF16><<<dim3(Dm / 128, T / 128), 256, 0, stream>>>(
        xn, WqT + (size_t)l * Dm * Dm, T, Dm, Dm, nullptr, qb, nullptr);
    gemm_kernel<EPI_BF16><<<dim3(Dm / 128, T / 128), 256, 0, stream>>>(
        xn, WkT + (size_t)l * Dm * Dm, T, Dm, Dm, nullptr, kb, nullptr);
    gemm_kernel<EPI_BF16><<<dim3(Dm / 128, T / 128), 256, 0, stream>>>(
        xn, WvT + (size_t)l * Dm * Dm, T, Dm, Dm, nullptr, vb, nullptr);

    rope_kernel<<<dim3(T, 2), 256, 0, stream>>>(qb);
    rope_kernel<<<dim3(T, 2), 256, 0, stream>>>(kb);

    attention_kernel<<<dim3(SEQ / 64, NH, BAT), 128, 0, stream>>>(qb, kb, vb, ab);

    gemm_kernel<EPI_RESID><<<dim3(Dm / 128, T / 128), 256, 0, stream>>>(
        ab, WoT + (size_t)l * Dm * Dm, T, Dm, Dm, x, nullptr, x);

    rmsnorm_kernel<<<dim3(T), 256, 0, stream>>>(x, n2w + (size_t)l * Dm, xn);

    gemm_kernel<EPI_F32><<<dim3((HID + 127) / 128, T / 128), 256, 0, stream>>>(
        xn, WgT + (size_t)l * HID * Dm, T, HID, Dm, gbuf, nullptr, nullptr);
    gemm_kernel<EPI_F32><<<dim3((HID + 127) / 128, T / 128), 256, 0, stream>>>(
        xn, WuT + (size_t)l * HID * Dm, T, HID, Dm, ubuf, nullptr, nullptr);

    {
      const size_t tot = (size_t)T * HIDp;
      silu_mul_kernel<<<dim3((unsigned)((tot + 255) / 256)), 256, 0, stream>>>(gbuf, ubuf, hp);
    }

    gemm_kernel<EPI_RESID><<<dim3(Dm / 128, T / 128), 256, 0, stream>>>(
        hp, WdT + (size_t)l * Dm * HIDp, T, Dm, HIDp, x, nullptr, x);
  }

  // --- final norm + logits ---
  rmsnorm_kernel<<<dim3(T), 256, 0, stream>>>(x, fnw, xn);
  gemm_kernel<EPI_F32><<<dim3(VOC / 128, T / 128), 256, 0, stream>>>(
      xn, embB, T, VOC, Dm, logits, nullptr, nullptr);
}
